// VotePoseProposal__51488067944621
// MI455X (gfx1250) — matrix operand order repaired and checked by reading, not compiled, not run on hardware
//
#include <hip/hip_runtime.h>
#include <hip/hip_bf16.h>

typedef __attribute__((ext_vector_type(16))) _Float16 v16h;
typedef __attribute__((ext_vector_type(8)))  _Float16 v8h;
typedef __attribute__((ext_vector_type(8)))  float    v8f;
typedef unsigned int u32x4 __attribute__((ext_vector_type(4)));
typedef int          i32x8 __attribute__((ext_vector_type(8)));
typedef int          i32x4 __attribute__((ext_vector_type(4)));

#define BKN   38          // B*K = 2*19
#define NPTS  1024        // M
#define NP    128         // NPOINT
#define NS    128         // NSAMPLE
#define K1P   288         // 259 padded to multiple of 32
#define KSTR1 296         // layer-1 LDS row stride (288 + 8 pad halves)
#define KSTR2 136         // 128 + 8 pad halves (conflict-free ds_load_b128)
#define RAD2  0.0225f
#define INVR  (1.0f/0.15f)

// ---------------- WMMA fragment loader --------------------------------------
__device__ __forceinline__ v16h load_frag16(const _Float16* p0, const _Float16* p1) {
    v16h r;
    *(v8h*)&r       = *(const v8h*)p0;   // elements 0..7
    *((v8h*)&r + 1) = *(const v8h*)p1;   // elements 8..15
    return r;
}

// ---------------- 128x128 = W(128xKPAD) * X(KPADx128) with BN+ReLU ----------
// Xin rows are KSTR-strided (padded) in LDS; K loop covers KPAD.
// MODE 0: write f16 to Hout (LDS, [n][KSTR2] transposed for next layer)
// MODE 1: per-channel max over samples -> atomicMax into featmax[128] (LDS)
// MODE 2: write f32 to gout ([o][n], stride 128) in global memory
template<int KPAD, int KSTR, int MODE>
__device__ __forceinline__ void gemm128(
    const _Float16* __restrict__ W,      // [128][KPAD] f16 (global, unpadded)
    const float* __restrict__ bb, const float* __restrict__ gg,
    const float* __restrict__ be,
    const _Float16* __restrict__ Xin,    // [128][KSTR] f16 (LDS, transposed)
    _Float16* Hout, float* featmax, float* gout, unsigned tid)
{
    const unsigned lane = tid & 31u, wave = tid >> 5;
    const unsigned lg = lane >> 4, ln = lane & 15u;
    const unsigned arow = wave * 16u + ln;

    v8f z = {0.f,0.f,0.f,0.f,0.f,0.f,0.f,0.f};
    v8f acc[8] = {z,z,z,z,z,z,z,z};

    for (int kc = 0; kc < KPAD; kc += 32) {
        const _Float16* wp = W + arow * KPAD + kc + lg * 8;
        v16h a = load_frag16(wp, wp + 16);
        v16h bf[8];
#pragma unroll
        for (int nt = 0; nt < 8; ++nt) {
            const _Float16* xp = Xin + (nt * 16 + ln) * KSTR + kc + lg * 8;
            bf[nt] = load_frag16(xp, xp + 16);
        }
#pragma unroll
        for (int nt = 0; nt < 8; ++nt)
            acc[nt] = __builtin_amdgcn_wmma_f32_16x16x32_f16(
                false, a, false, bf[nt], (short)0, acc[nt], false, false);
    }

    const unsigned ob = wave * 16u + lg * 8u;      // first of 8 output channels
    float sc[8], sb[8], sbe[8];
    const float inv = rsqrtf(1.0f + 1e-5f);
#pragma unroll
    for (int j = 0; j < 8; ++j) {
        sc[j]  = gg[ob + j] * inv;
        sb[j]  = bb[ob + j];
        sbe[j] = be[ob + j];
    }
    float mx[8] = {0.f,0.f,0.f,0.f,0.f,0.f,0.f,0.f};
#pragma unroll
    for (int nt = 0; nt < 8; ++nt) {
        unsigned n = nt * 16 + ln;
        if constexpr (MODE == 0) {
            union { _Float16 h[8]; uint4 u; } pk;
#pragma unroll
            for (int j = 0; j < 8; ++j) {
                float v = fmaxf(0.f, (acc[nt][j] + sb[j]) * sc[j] + sbe[j]);
                pk.h[j] = (_Float16)v;
            }
            *(uint4*)(Hout + n * KSTR2 + ob) = pk.u;
        } else if constexpr (MODE == 2) {
#pragma unroll
            for (int j = 0; j < 8; ++j) {
                float v = fmaxf(0.f, (acc[nt][j] + sb[j]) * sc[j] + sbe[j]);
                gout[(ob + j) * 128 + n] = v;
            }
        } else {
#pragma unroll
            for (int j = 0; j < 8; ++j) {
                float v = fmaxf(0.f, (acc[nt][j] + sb[j]) * sc[j] + sbe[j]);
                mx[j] = fmaxf(mx[j], v);
            }
        }
    }
    if constexpr (MODE == 1) {
#pragma unroll
        for (int j = 0; j < 8; ++j)
            atomicMax((int*)&featmax[ob + j], __float_as_int(mx[j]));
    }
}

// ---------------- weight conversion to f16 (with K padding for layer 1) -----
__global__ void k_cvt(const float* __restrict__ m1w, const float* __restrict__ m2w,
                      const float* __restrict__ m3w, const float* __restrict__ c1w,
                      const float* __restrict__ c2w,
                      _Float16* W1, _Float16* W2, _Float16* W3,
                      _Float16* C1, _Float16* C2)
{
    int i = blockIdx.x * blockDim.x + threadIdx.x;
    if (i < 128 * K1P) {
        int r = i / K1P, c = i - r * K1P;
        W1[i] = (_Float16)(c < 259 ? m1w[r * 259 + c] : 0.f);
    }
    if (i < 128 * 128) {
        W2[i] = (_Float16)m2w[i];
        W3[i] = (_Float16)m3w[i];
        C1[i] = (_Float16)c1w[i];
        C2[i] = (_Float16)c2w[i];
    }
}

// ---------------- farthest point sampling (one block per batch) -------------
__global__ void __launch_bounds__(256) k_fps(const float* __restrict__ xyz,
                                             float* __restrict__ nxyz,
                                             float* __restrict__ aggout)
{
    __shared__ float xl[NPTS * 3];
    __shared__ float dist[NPTS];
    __shared__ float sval[256];
    __shared__ int   sidr[256];
    __shared__ int   sfi[NP];
    __shared__ int   sfar;
    int bk = blockIdx.x, b = bk / 19, kk = bk - b * 19;
    int tid = threadIdx.x;
    for (int m = tid; m < NPTS; m += 256) {
#pragma unroll
        for (int d = 0; d < 3; ++d)
            xl[m * 3 + d] = xyz[((b * NPTS + m) * 19 + kk) * 3 + d];
        dist[m] = 1e10f;
    }
    if (tid == 0) sfar = 0;
    __syncthreads();
    for (int it = 0; it < NP; ++it) {
        int cf = sfar;
        if (tid == 0) sfi[it] = cf;
        float cx = xl[cf * 3], cy = xl[cf * 3 + 1], cz = xl[cf * 3 + 2];
        float bv = -1.f; int bi = 0;
        for (int m = tid; m < NPTS; m += 256) {
            float dx = xl[m * 3] - cx, dy = xl[m * 3 + 1] - cy, dz = xl[m * 3 + 2] - cz;
            float d = dx * dx + dy * dy + dz * dz;
            float nd = fminf(dist[m], d);
            dist[m] = nd;
            if (nd > bv) { bv = nd; bi = m; }
        }
        sval[tid] = bv; sidr[tid] = bi;
        __syncthreads();
        for (int s = 128; s > 0; s >>= 1) {
            if (tid < s) {
                float ov = sval[tid + s]; int oi = sidr[tid + s];
                if (ov > sval[tid] || (ov == sval[tid] && oi < sidr[tid])) {
                    sval[tid] = ov; sidr[tid] = oi;
                }
            }
            __syncthreads();
        }
        if (tid == 0) sfar = sidr[0];
        __syncthreads();
    }
    for (int e = tid; e < NP * 3; e += 256) {
        int p = e / 3, d = e - p * 3;
        float v = xl[sfi[p] * 3 + d];
        nxyz[(bk * NP + p) * 3 + d]   = v;
        aggout[(bk * NP + p) * 3 + d] = v;
    }
}

// ---------------- ball query: first NS in-radius indices, ascending ---------
__global__ void __launch_bounds__(128) k_ball(const float* __restrict__ xyz,
                                              const float* __restrict__ nxyz,
                                              int* __restrict__ gidx)
{
    __shared__ float xl[NPTS * 3];
    int bk = blockIdx.x, b = bk / 19, kk = bk - b * 19;
    int tid = threadIdx.x;
    for (int m = tid; m < NPTS; m += 128) {
#pragma unroll
        for (int d = 0; d < 3; ++d)
            xl[m * 3 + d] = xyz[((b * NPTS + m) * 19 + kk) * 3 + d];
    }
    __syncthreads();
    int p = tid;
    float cx = nxyz[(bk * NP + p) * 3];
    float cy = nxyz[(bk * NP + p) * 3 + 1];
    float cz = nxyz[(bk * NP + p) * 3 + 2];
    long base = (long)(bk * NP + p) * NS;
    int cnt = 0, first = 0;
    for (int m = 0; m < NPTS; ++m) {
        float dx = xl[m * 3] - cx, dy = xl[m * 3 + 1] - cy, dz = xl[m * 3 + 2] - cz;
        float d2 = dx * dx + dy * dy + dz * dz;
        if (d2 < RAD2) {
            if (cnt == 0) first = m;
            if (cnt < NS) gidx[base + cnt] = m;
            cnt++;
        }
    }
    if (cnt > NS) cnt = NS;
    for (int s = cnt; s < NS; ++s) gidx[base + s] = first;
}

// ---------------- fused gather + 3-layer WMMA MLP + sample-max --------------
// grid (NP, BKN), 256 threads. Dyn LDS = 111632 B.
__global__ void __launch_bounds__(256) k_mlp(
    const float* __restrict__ xyz, const float* __restrict__ feats,
    const int* __restrict__ gidx, const float* __restrict__ nxyz,
    const _Float16* __restrict__ W1, const _Float16* __restrict__ W2,
    const _Float16* __restrict__ W3,
    const float* m1b, const float* m1g, const float* m1be,
    const float* m2b, const float* m2g, const float* m2be,
    const float* m3b, const float* m3g, const float* m3be,
    float* __restrict__ featout, _Float16* __restrict__ featH)
{
    extern __shared__ char smem[];
    _Float16* XT   = (_Float16*)smem;                 // [128][296] = 75776 B
    _Float16* HT   = (_Float16*)(smem + 75776);       // [128][136] = 34816 B
    int*      sidx = (int*)  (smem + 110592);         // 512 B
    float*    scen = (float*)(smem + 111104);         // 16 B
    float*    smax = (float*)(smem + 111120);         // 512 B
    int bk = blockIdx.y, p = blockIdx.x;
    int b = bk / 19, kk = bk - b * 19;
    unsigned tid = threadIdx.x;

    if (tid < NP) sidx[tid] = gidx[(long)(bk * NP + p) * NS + tid];
    if (tid < 3)  scen[tid] = nxyz[(bk * NP + p) * 3 + tid];
    if (tid < NP) smax[tid] = 0.f;
    // warm L2/WGP$ with next layers' weights while we gather + run layer 1
    __builtin_prefetch((const char*)W2 + tid * 128, 0, 0);
    __builtin_prefetch((const char*)W3 + tid * 128, 0, 0);
    __syncthreads();

    { // gather layer-1 input, transposed: XT[n][c] (row stride KSTR1)
        int n = tid >> 1, half = tid & 1;
        int id = sidx[n];
        long fbase = ((long)(b * 256) * 19 + kk) * 1024 + id;
        for (int c = half; c < K1P; c += 2) {
            float v;
            if (c < 3)        v = (xyz[((b * NPTS + id) * 19 + kk) * 3 + c] - scen[c]) * INVR;
            else if (c < 259) v = feats[fbase + (long)(c - 3) * 19456];
            else              v = 0.f;
            XT[n * KSTR1 + c] = (_Float16)v;
        }
    }
    __syncthreads();
    gemm128<K1P, KSTR1, 0>(W1, m1b, m1g, m1be, XT, HT, nullptr, nullptr, tid);
    __syncthreads();
    gemm128<128, KSTR2, 0>(W2, m2b, m2g, m2be, HT, XT, nullptr, nullptr, tid);
    __syncthreads();
    gemm128<128, KSTR2, 1>(W3, m3b, m3g, m3be, XT, nullptr, smax, nullptr, tid);
    __syncthreads();
    if (tid < NP) {
        float fv = smax[tid];
        featout[((long)bk * 128 + tid) * 128 + p] = fv;             // feat (bk,o,p)
        featH[(long)(bk * NP + p) * 128 + tid] = (_Float16)fv;      // transposed f16
    }
}

// ---------------- fused c1+c2 WMMA head, one block per batch ----------------
// TDM stages the 128x128 f16 feat tile into LDS, using the D# pad feature to
// land rows at the conflict-free KSTR2 stride (64-dword interval, 4-dword pad).
__global__ void __launch_bounds__(256) k_head(
    const _Float16* __restrict__ featH,
    const _Float16* __restrict__ C1, const _Float16* __restrict__ C2,
    const float* c1b, const float* c1g, const float* c1be,
    const float* c2b, const float* c2g, const float* c2be,
    float* __restrict__ netF)
{
    extern __shared__ char smem[];
    _Float16* XT = (_Float16*)smem;            // [128][136] = 34816 B
    _Float16* HT = (_Float16*)(smem + 34816);  // [128][136] = 34816 B
    int bk = blockIdx.x;
    unsigned tid = threadIdx.x;

    // ---- Tensor Data Mover: global (128x128 f16, stride 128) -> LDS @0 ----
    unsigned long ga = (unsigned long)(const void*)(featH + (long)bk * 16384);
    u32x4 g0;
    g0[0] = 1u;                                   // count=1, user D#
    g0[1] = 0u;                                   // lds_addr = 0 (XT @ dyn base)
    g0[2] = (unsigned)ga;                         // global_addr[31:0]
    g0[3] = (unsigned)(ga >> 32) | (2u << 30);    // global_addr[56:32], type=2
    i32x8 g1;
    g1[0] = (int)((1u << 16) | (1u << 20) | (5u << 22) | (3u << 25));
    //        data_size=2B    pad_enable    interval=64dw   amount=4dw
    g1[1] = (int)(128u << 16);                    // tensor_dim0 = 128
    g1[2] = (int)(128u << 16);                    // tensor_dim1 = 128
    g1[3] = (int)(128u << 16);                    // tile_dim0 = 128
    g1[4] = 128;                                  // tile_dim1 = 128
    g1[5] = 128;                                  // tensor_dim0_stride = 128
    g1[6] = 0; g1[7] = 0;
    i32x4 gz4 = {0, 0, 0, 0};                     // groups 2/3 unused (2-D)
    i32x8 gz8 = {0, 0, 0, 0, 0, 0, 0, 0};
    if (tid < 32) {                               // single wave issues the DMA
        __builtin_amdgcn_tensor_load_to_lds(g0, g1, gz4, gz4, gz8, 0);
        __builtin_amdgcn_s_wait_tensorcnt(0);
    }
    __syncthreads();

    gemm128<128, KSTR2, 0>(C1, c1b, c1g, c1be, XT, HT, nullptr, nullptr, tid);
    __syncthreads();
    gemm128<128, KSTR2, 2>(C2, c2b, c2g, c2be, HT, nullptr, nullptr,
                           netF + (long)bk * 16384, tid);
}

// ---------------- op einsum + output assembly -------------------------------
__global__ void __launch_bounds__(128) k_final(
    const float* __restrict__ netF, const float* __restrict__ opw,
    const float* __restrict__ opb, float* __restrict__ scores,
    const float* __restrict__ agg)
{
    __shared__ float sw[5 * 128];
    __shared__ float sb5[5];
    int bk = blockIdx.x, kk = bk - (bk / 19) * 19;
    int tid = threadIdx.x;
    for (int i = tid; i < 640; i += 128) sw[i] = opw[kk * 640 + i];
    if (tid < 5) sb5[tid] = opb[kk * 5 + tid];
    __syncthreads();
    int p = tid;
    float a0 = 0, a1 = 0, a2 = 0, a3 = 0, a4 = 0;
    const float* nb = netF + (long)bk * 16384 + p;
    for (int c = 0; c < 128; ++c) {
        float v = nb[c * 128];
        a0 += sw[c] * v; a1 += sw[128 + c] * v; a2 += sw[256 + c] * v;
        a3 += sw[384 + c] * v; a4 += sw[512 + c] * v;
    }
    a0 += sb5[0]; a1 += sb5[1]; a2 += sb5[2]; a3 += sb5[3]; a4 += sb5[4];
    long so = (long)(bk * NP + p) * 5;
    const float* ag = agg + (long)(bk * NP + p) * 3;
    scores[so + 0] = a0;
    scores[so + 1] = a1;
    scores[so + 2] = ag[0] + a2;
    scores[so + 3] = ag[1] + a3;
    scores[so + 4] = ag[2] + a4;
}

extern "C" void kernel_launch(void* const* d_in, const int* in_sizes, int n_in,
                              void* d_out, int out_size, void* d_ws, size_t ws_size,
                              hipStream_t stream) {
    (void)in_sizes; (void)n_in; (void)out_size; (void)ws_size;
    const float* xyz   = (const float*)d_in[0];
    const float* feats = (const float*)d_in[1];
    const float* m1w = (const float*)d_in[2],  *m1b = (const float*)d_in[3];
    const float* m1g = (const float*)d_in[4],  *m1be = (const float*)d_in[5];
    const float* m2w = (const float*)d_in[6],  *m2b = (const float*)d_in[7];
    const float* m2g = (const float*)d_in[8],  *m2be = (const float*)d_in[9];
    const float* m3w = (const float*)d_in[10], *m3b = (const float*)d_in[11];
    const float* m3g = (const float*)d_in[12], *m3be = (const float*)d_in[13];
    const float* c1w = (const float*)d_in[14], *c1b = (const float*)d_in[15];
    const float* c1g = (const float*)d_in[16], *c1be = (const float*)d_in[17];
    const float* c2w = (const float*)d_in[18], *c2b = (const float*)d_in[19];
    const float* c2g = (const float*)d_in[20], *c2be = (const float*)d_in[21];
    const float* opw = (const float*)d_in[22], *opb = (const float*)d_in[23];

    float* out    = (float*)d_out;
    float* scores = out;                  // 2*19*128*5 = 24320
    float* aggo   = out + 24320;          // 2*19*128*3 = 14592
    float* feato  = out + 38912;          // 2*19*128*128

    char* ws = (char*)d_ws;
    _Float16* W1    = (_Float16*)(ws);             // 73728 B
    _Float16* W2    = (_Float16*)(ws + 73728);     // 32768 B
    _Float16* W3    = (_Float16*)(ws + 106496);    // 32768 B
    _Float16* C1    = (_Float16*)(ws + 139264);    // 32768 B
    _Float16* C2    = (_Float16*)(ws + 172032);    // 32768 B
    float*    nxyz  = (float*)  (ws + 204800);     // 58368 B
    int*      gidx  = (int*)    (ws + 263168);     // 2490368 B
    _Float16* featH = (_Float16*)(ws + 2753536);   // 1245184 B
    float*    netF  = (float*)  (ws + 3998720);    // 2490368 B -> 6489088 total

    k_cvt<<<144, 256, 0, stream>>>(m1w, m2w, m3w, c1w, c2w, W1, W2, W3, C1, C2);
    k_fps<<<BKN, 256, 0, stream>>>(xyz, nxyz, aggo);
    k_ball<<<BKN, 128, 0, stream>>>(xyz, nxyz, gidx);
    dim3 g3(NP, BKN);
    k_mlp<<<g3, 256, 111632, stream>>>(xyz, feats, gidx, nxyz, W1, W2, W3,
                                       m1b, m1g, m1be, m2b, m2g, m2be,
                                       m3b, m3g, m3be, feato, featH);
    k_head<<<BKN, 256, 69632, stream>>>(featH, C1, C2, c1b, c1g, c1be,
                                        c2b, c2g, c2be, netF);
    k_final<<<BKN, 128, 0, stream>>>(netF, opw, opb, scores, aggo);
}
